// FCLModule_74131135529089
// MI455X (gfx1250) — matrix operand order repaired and checked by reading, not compiled, not run on hardware
//
#include <hip/hip_runtime.h>
#include <hip/hip_bf16.h>

// ---------------------------------------------------------------------------
// MI455X (gfx1250) fused MoE-MLP + LayerNorm, WMMA f16 (f32 accumulate).
//
// Pipeline per 16-row tile (one wave32 per workgroup):
//   h1   = relu(x*w1+b1)            per-lane VALU (K=32, built directly as A-frag)
//   h2   = relu(h1 @ w2 + b2)       WMMA 16x16x32, both experts
//   t    = relu(h2 @ w3 + b3)       WMMA, both experts, per-row select -> x1
//   x2   = relu(x1 @ s2w + s2b)+x1  WMMA + residual
//   h    = relu(x2 @ w3a + b3a)     WMMA
//   h    = h @ w3b + b3b            WMMA (f32 kept in LDS)
//   out  = LayerNorm(h)*g + b       f32, shfl_xor(16) row reduce, float4 stores
//
// Weights are pre-packed f32->f16 into d_ws in WMMA-B fragment layout:
//   frag = 1KB = 32 lanes x 16 halfs; lane holds column N = lane&15,
//   K = ((lane&16)?8:0) + ((t&8)?16:0) + (t&7)  (two runs of 8 K per lane).
// A fragments (16-bit A 16x32 layout per ISA 7.12.2) rebuilt from LDS with
// two ds_load_b128 per fragment.
// ---------------------------------------------------------------------------

typedef __attribute__((ext_vector_type(16))) _Float16 v16h;
typedef __attribute__((ext_vector_type(8)))  float    v8f;

union V16H { v16h v; _Float16 h[16]; uint32_t u[8]; };

#define TILE_M     16
#define STRIDE_S   264   // f16 stride, 256-wide buffer (+8 pad) -> 528B rows, 16B aligned
#define STRIDE_L   520   // f16 stride, 512-wide buffer (+8 pad) -> 1040B rows, 16B aligned
#define STRIDE_H   516   // f32 stride for LN buffer -> 2064B rows, 16B aligned
#define LN_EPS     1e-5f

// fragment-region offsets inside d_ws (units: fragments of 512 halfs)
#define OFF_C2   0      // cw2: 32x64    -> KT=1,  NT=4   (4)
#define OFF_R2   4      // rw2                            (4)
#define OFF_C3   8      // cw3: 64x256   -> KT=2,  NT=16  (32)
#define OFF_R3   40     // rw3                            (32)
#define OFF_S2   72     // s2w: 256x256  -> KT=8,  NT=16  (128)
#define OFF_3A   200    // w3a: 256x512  -> KT=8,  NT=32  (256)
#define OFF_3B   456    // w3b: 512x512  -> KT=16, NT=32  (512)
#define N_FRAGS  968

__device__ __forceinline__ v8f wmma32f16(v16h a, v16h b, v8f c) {
  // D = A(16x32 f16) x B(32x16 f16) + C(16x16 f32)
  return __builtin_amdgcn_wmma_f32_16x16x32_f16(false, a, false, b,
                                                (short)0, c, false, false);
}

__device__ __forceinline__ v8f splat8(float x) {
  v8f c;
#pragma unroll
  for (int i = 0; i < 8; ++i) c[i] = x;
  return c;
}

// A fragment (16x32 f16) from LDS row-major activations.
// ISA layout: lane m=lane&15; kh=(lane&16)?8:0; VGPR0..3 = K kh+0..7 (pairs),
// VGPR4..7 = K kh+16..23  -> two contiguous 16B loads per lane.
__device__ __forceinline__ v16h lds_a_frag(const _Float16* base, int stride,
                                           int kbase, int lane) {
  const int m  = lane & 15;
  const int kh = (lane & 16) ? 8 : 0;
  const _Float16* p = base + m * stride + kbase + kh;
  V16H a;
  *(uint4*)(&a.u[0]) = *(const uint4*)(p);
  *(uint4*)(&a.u[4]) = *(const uint4*)(p + 16);
  return a.v;
}

// B fragment (32x16 f16) from pre-packed global: 32B contiguous per lane.
__device__ __forceinline__ v16h g_b_frag(const _Float16* wp, int frag, int lane) {
  const _Float16* p = wp + ((size_t)frag << 9) + (lane << 4);
  V16H b;
  *(uint4*)(&b.u[0]) = *(const uint4*)(p);
  *(uint4*)(&b.u[4]) = *(const uint4*)(p + 8);
  return b.v;
}

// ---------------------------------------------------------------------------
// Weight pack: f32 row-major W[K][N] -> f16 B-fragments.
// fragment f: kt = f % KT, nt = f / KT; element e: lane=e>>4, t=e&15;
//   N = nt*16 + (lane&15);  K = kt*32 + ((lane&16)>>1) + ((t&8)<<1) + (t&7)
// ---------------------------------------------------------------------------
__global__ void pack_b_kernel(const float* __restrict__ W,
                              _Float16* __restrict__ dst, int K, int N) {
  const int KT = K >> 5;
  const int f  = blockIdx.x;
  const int kt = f % KT, nt = f / KT;
  for (int e = threadIdx.x; e < 512; e += blockDim.x) {
    const int lane = e >> 4;
    const int t    = e & 15;
    const int k    = kt * 32 + ((lane & 16) >> 1) + ((t & 8) << 1) + (t & 7);
    const int n    = nt * 16 + (lane & 15);
    dst[(size_t)f * 512 + e] = (_Float16)W[(size_t)k * N + n];
  }
}

// ---------------------------------------------------------------------------
// Fused pipeline kernel: grid = B/16 blocks x 32 threads (one wave each).
// ---------------------------------------------------------------------------
__global__ void __launch_bounds__(32)
fused_mlp_ln_kernel(const float* __restrict__ x,
                    const int*   __restrict__ stype,
                    const float* __restrict__ cw1, const float* __restrict__ cb1,
                    const float* __restrict__ rw1, const float* __restrict__ rb1,
                    const float* __restrict__ cb2, const float* __restrict__ rb2,
                    const float* __restrict__ cb3, const float* __restrict__ rb3,
                    const float* __restrict__ s2b,
                    const float* __restrict__ b3a, const float* __restrict__ b3b,
                    const float* __restrict__ lng, const float* __restrict__ lnb,
                    const _Float16* __restrict__ wp,
                    float* __restrict__ out) {
  __shared__ __align__(16) _Float16 sBufS[TILE_M * STRIDE_S]; // 256-wide
  __shared__ __align__(16) _Float16 sBufL[TILE_M * STRIDE_L]; // 512-wide
  __shared__ __align__(16) float    sH[TILE_M * STRIDE_H];    // f32 LN buffer
  __shared__ int   sSt[TILE_M];
  __shared__ float sMu[TILE_M], sRs[TILE_M];

  const int lane = threadIdx.x & 31;
  const int m    = lane & 15;           // row (A/layer1) or column (D stores)
  const int mrb  = (lane & 16) ? 8 : 0; // D-fragment row base
  const size_t rowBase = (size_t)blockIdx.x * TILE_M;

  const _Float16* wpC2 = wp + (size_t)OFF_C2 * 512;
  const _Float16* wpR2 = wp + (size_t)OFF_R2 * 512;
  const _Float16* wpC3 = wp + (size_t)OFF_C3 * 512;
  const _Float16* wpR3 = wp + (size_t)OFF_R3 * 512;
  const _Float16* wpS2 = wp + (size_t)OFF_S2 * 512;
  const _Float16* wp3A = wp + (size_t)OFF_3A * 512;
  const _Float16* wp3B = wp + (size_t)OFF_3B * 512;

  // ---- layer 1: h1 = relu(x*w1+b1), built directly as A fragments ----------
  const float xv = x[rowBase + m];
  const int   st = stype[rowBase + m];
  if (lane < TILE_M) sSt[lane] = st;

  V16H a1c, a1r;
  const int kh = (lane & 16) ? 8 : 0;
#pragma unroll
  for (int t = 0; t < 16; ++t) {
    const int k = kh + ((t & 8) << 1) + (t & 7); // 16-bit A-layout K index
    a1c.h[t] = (_Float16)fmaxf(xv * cw1[k] + cb1[k], 0.0f);
    a1r.h[t] = (_Float16)fmaxf(xv * rw1[k] + rb1[k], 0.0f);
  }
  __syncthreads();

  int sel[8];
#pragma unroll
  for (int r = 0; r < 8; ++r) sel[r] = sSt[mrb + r];

  // ---- layer 2: 32 -> 64, both experts (h2_c cols 0..63, h2_r cols 64..127)
  for (int nt = 0; nt < 4; ++nt) {
    v8f cc = splat8(cb2[nt * 16 + m]);
    cc = wmma32f16(a1c.v, g_b_frag(wpC2, nt, lane), cc);
    v8f cr = splat8(rb2[nt * 16 + m]);
    cr = wmma32f16(a1r.v, g_b_frag(wpR2, nt, lane), cr);
#pragma unroll
    for (int r = 0; r < 8; ++r) {
      sBufS[(mrb + r) * STRIDE_S + nt * 16 + m]      = (_Float16)fmaxf(cc[r], 0.0f);
      sBufS[(mrb + r) * STRIDE_S + 64 + nt * 16 + m] = (_Float16)fmaxf(cr[r], 0.0f);
    }
  }
  __syncthreads();

  // ---- layer 3: 64 -> 256, both experts, per-row expert select -> x1 -------
  v16h a2c0 = lds_a_frag(sBufS,      STRIDE_S, 0,  lane);
  v16h a2c1 = lds_a_frag(sBufS,      STRIDE_S, 32, lane);
  v16h a2r0 = lds_a_frag(sBufS + 64, STRIDE_S, 0,  lane);
  v16h a2r1 = lds_a_frag(sBufS + 64, STRIDE_S, 32, lane);
  for (int nt = 0; nt < 16; ++nt) {
    v8f cc = splat8(cb3[nt * 16 + m]);
    cc = wmma32f16(a2c0, g_b_frag(wpC3, nt * 2 + 0, lane), cc);
    cc = wmma32f16(a2c1, g_b_frag(wpC3, nt * 2 + 1, lane), cc);
    v8f cr = splat8(rb3[nt * 16 + m]);
    cr = wmma32f16(a2r0, g_b_frag(wpR3, nt * 2 + 0, lane), cr);
    cr = wmma32f16(a2r1, g_b_frag(wpR3, nt * 2 + 1, lane), cr);
#pragma unroll
    for (int r = 0; r < 8; ++r) {
      const float vc = fmaxf(cc[r], 0.0f), vr = fmaxf(cr[r], 0.0f);
      sBufL[(mrb + r) * STRIDE_L + nt * 16 + m] = (_Float16)(sel[r] ? vr : vc);
    }
  }
  __syncthreads();

  // ---- stage 2: x2 = relu(x1 @ s2w + s2b) + x1 -----------------------------
  v16h ax[8];
#pragma unroll
  for (int kt = 0; kt < 8; ++kt) ax[kt] = lds_a_frag(sBufL, STRIDE_L, kt * 32, lane);
  for (int nt = 0; nt < 16; ++nt) {
    v8f acc = splat8(s2b[nt * 16 + m]);
#pragma unroll
    for (int kt = 0; kt < 8; ++kt)
      acc = wmma32f16(ax[kt], g_b_frag(wpS2, nt * 8 + kt, lane), acc);
#pragma unroll
    for (int r = 0; r < 8; ++r) {
      const float x1v = (float)sBufL[(mrb + r) * STRIDE_L + nt * 16 + m];
      sBufS[(mrb + r) * STRIDE_S + nt * 16 + m] =
          (_Float16)(fmaxf(acc[r], 0.0f) + x1v);
    }
  }
  __syncthreads();

  // ---- stage 3a: 256 -> 512, relu ------------------------------------------
#pragma unroll
  for (int kt = 0; kt < 8; ++kt) ax[kt] = lds_a_frag(sBufS, STRIDE_S, kt * 32, lane);
  for (int nt = 0; nt < 32; ++nt) {
    v8f acc = splat8(b3a[nt * 16 + m]);
#pragma unroll
    for (int kt = 0; kt < 8; ++kt)
      acc = wmma32f16(ax[kt], g_b_frag(wp3A, nt * 8 + kt, lane), acc);
#pragma unroll
    for (int r = 0; r < 8; ++r)
      sBufL[(mrb + r) * STRIDE_L + nt * 16 + m] = (_Float16)fmaxf(acc[r], 0.0f);
  }
  __syncthreads();

  // ---- stage 3b: 512 -> 512 (no relu), f32 into LDS for LayerNorm ----------
  v16h ay[16];
#pragma unroll
  for (int kt = 0; kt < 16; ++kt) ay[kt] = lds_a_frag(sBufL, STRIDE_L, kt * 32, lane);
  for (int nt = 0; nt < 32; ++nt) {
    v8f acc = splat8(b3b[nt * 16 + m]);
#pragma unroll
    for (int kt = 0; kt < 16; ++kt)
      acc = wmma32f16(ay[kt], g_b_frag(wp3B, nt * 16 + kt, lane), acc);
#pragma unroll
    for (int r = 0; r < 8; ++r)
      sH[(mrb + r) * STRIDE_H + nt * 16 + m] = acc[r];
  }
  __syncthreads();

  // ---- LayerNorm: stats (each half-wave lane owns 256 cols of one row) -----
  {
    const int   row  = lane & 15;
    const int   hsel = lane >> 4;
    const float* hp  = sH + row * STRIDE_H + hsel * 256;
    float s = 0.0f, s2 = 0.0f;
#pragma unroll 8
    for (int i = 0; i < 256; i += 4) {
      const float4 v4 = *(const float4*)(hp + i);
      s  += v4.x + v4.y + v4.z + v4.w;
      s2 += v4.x * v4.x + v4.y * v4.y + v4.z * v4.z + v4.w * v4.w;
    }
    s  += __shfl_xor(s, 16);
    s2 += __shfl_xor(s2, 16);
    const float mu  = s * (1.0f / 512.0f);
    const float var = s2 * (1.0f / 512.0f) - mu * mu;
    const float rs  = rsqrtf(var + LN_EPS);
    if (lane < TILE_M) { sMu[lane] = mu; sRs[lane] = rs; }
  }
  __syncthreads();

  // ---- normalize + affine + coalesced float4 store (one row / iteration) ---
  float4 g4[4], b4[4];
#pragma unroll
  for (int it = 0; it < 4; ++it) {
    const int c = it * 128 + lane * 4;
    g4[it] = *(const float4*)(lng + c);
    b4[it] = *(const float4*)(lnb + c);
  }
  for (int mr = 0; mr < TILE_M; ++mr) {
    const float mu = sMu[mr], rs = sRs[mr];
    float* op = out + (rowBase + mr) * 512;
#pragma unroll
    for (int it = 0; it < 4; ++it) {
      const int c = it * 128 + lane * 4;
      const float4 h4 = *(const float4*)(sH + mr * STRIDE_H + c);
      float4 o;
      o.x = (h4.x - mu) * rs * g4[it].x + b4[it].x;
      o.y = (h4.y - mu) * rs * g4[it].y + b4[it].y;
      o.z = (h4.z - mu) * rs * g4[it].z + b4[it].z;
      o.w = (h4.w - mu) * rs * g4[it].w + b4[it].w;
      *(float4*)(op + c) = o;
    }
  }
}

extern "C" void kernel_launch(void* const* d_in, const int* in_sizes, int n_in,
                              void* d_out, int out_size, void* d_ws, size_t ws_size,
                              hipStream_t stream) {
  const float* x   = (const float*)d_in[0];
  const int*   st  = (const int*)  d_in[1];
  const float* cw1 = (const float*)d_in[2];  const float* cb1 = (const float*)d_in[3];
  const float* cw2 = (const float*)d_in[4];  const float* cb2 = (const float*)d_in[5];
  const float* cw3 = (const float*)d_in[6];  const float* cb3 = (const float*)d_in[7];
  const float* rw1 = (const float*)d_in[8];  const float* rb1 = (const float*)d_in[9];
  const float* rw2 = (const float*)d_in[10]; const float* rb2 = (const float*)d_in[11];
  const float* rw3 = (const float*)d_in[12]; const float* rb3 = (const float*)d_in[13];
  const float* s2w = (const float*)d_in[14]; const float* s2b = (const float*)d_in[15];
  const float* w3a = (const float*)d_in[16]; const float* b3a = (const float*)d_in[17];
  const float* w3b = (const float*)d_in[18]; const float* b3b = (const float*)d_in[19];
  const float* lng = (const float*)d_in[20]; const float* lnb = (const float*)d_in[21];
  (void)n_in; (void)out_size; (void)ws_size;

  _Float16* wp = (_Float16*)d_ws;

  // pack weights into WMMA B-fragment f16 layout (deterministic every call)
  pack_b_kernel<<<4,   256, 0, stream>>>(cw2, wp + (size_t)OFF_C2 * 512, 32,  64);
  pack_b_kernel<<<4,   256, 0, stream>>>(rw2, wp + (size_t)OFF_R2 * 512, 32,  64);
  pack_b_kernel<<<32,  256, 0, stream>>>(cw3, wp + (size_t)OFF_C3 * 512, 64,  256);
  pack_b_kernel<<<32,  256, 0, stream>>>(rw3, wp + (size_t)OFF_R3 * 512, 64,  256);
  pack_b_kernel<<<128, 256, 0, stream>>>(s2w, wp + (size_t)OFF_S2 * 512, 256, 256);
  pack_b_kernel<<<256, 256, 0, stream>>>(w3a, wp + (size_t)OFF_3A * 512, 256, 512);
  pack_b_kernel<<<512, 256, 0, stream>>>(w3b, wp + (size_t)OFF_3B * 512, 512, 512);

  const int nrows = in_sizes[0]; // B = 262144
  fused_mlp_ln_kernel<<<nrows / TILE_M, 32, 0, stream>>>(
      x, st, cw1, cb1, rw1, rb1, cb2, rb2, cb3, rb3,
      s2b, b3a, b3b, lng, lnb, wp, (float*)d_out);
}
